// ParallelDirectPoincareDynamics_28252294873664
// MI455X (gfx1250) — compile-verified, matrix operand order checked by reading, not run on hardware
//
#include <hip/hip_runtime.h>
#include <hip/hip_bf16.h>

// ---------------------------------------------------------------------------
// Problem constants (reference: B=4096, D=2048, T=16)
// ---------------------------------------------------------------------------
constexpr int Bn = 4096;
constexpr int Dn = 2048;
constexpr int Tn = 16;
constexpr int KSTEPS = Dn / 32;
constexpr float EPSF   = 1e-7f;
constexpr float MAXN   = 1.0f - 1e-5f;
constexpr float CLIPAT = 1.0f - 1e-5f;

typedef __attribute__((ext_vector_type(16))) __bf16 v16bf;
typedef __attribute__((ext_vector_type(8)))  float  v8f;

// ---------------------------------------------------------------------------
// Helpers
// ---------------------------------------------------------------------------
__device__ __forceinline__ unsigned short bf16_bits(float x) {
    __bf16 h = (__bf16)x;                       // RNE f32->bf16
    return __builtin_bit_cast(unsigned short, h);
}

// split two f32 into packed hi / lo bf16 dwords (bf16x2 error ~2^-17 rel.)
__device__ __forceinline__ void split2(float a, float b,
                                       unsigned int& hi, unsigned int& lo) {
    __bf16 ah = (__bf16)a, bh = (__bf16)b;
    float  ar = a - (float)ah, br = b - (float)bh;
    unsigned short ahi = __builtin_bit_cast(unsigned short, ah);
    unsigned short bhi = __builtin_bit_cast(unsigned short, bh);
    unsigned short alo = bf16_bits(ar);
    unsigned short blo = bf16_bits(br);
    hi = (unsigned int)ahi | ((unsigned int)bhi << 16);
    lo = (unsigned int)alo | ((unsigned int)blo << 16);
}

struct Pack32B { uint4 a, b; };  // 32 bytes == v16bf

// Load a 16x32 bf16 WMMA operand fragment from LDS.
// LDS layout: row-major, 32 bf16 (=16 dwords) per row.
// lane<16  : row = l15, K = {0..7, 16..23}
// lane>=16 : row = l15, K = {8..15, 24..31}
// base must already point at &lds[row*16 + half*4].
__device__ __forceinline__ v16bf load_frag(const unsigned int* base) {
    Pack32B p;
    p.a = *reinterpret_cast<const uint4*>(base);      // K  half*8 .. +7
    p.b = *reinterpret_cast<const uint4*>(base + 8);  // K  16+half*8 .. +7
    return __builtin_bit_cast(v16bf, p);
}

__device__ __forceinline__ v8f wmma_bf16(v16bf a, v16bf b, v8f c) {
    return __builtin_amdgcn_wmma_f32_16x16x32_bf16(
        /*neg_a=*/false, a, /*neg_b=*/false, b,
        /*c_mod=*/(short)0, c, /*reuse_a=*/false, /*reuse_b=*/false);
}

// ---------------------------------------------------------------------------
// Kernel 1: mv = v_init @ W^T  (bf16x3 split, f32 accumulate)
//   WG tile: 128(M) x 128(N); 8 waves as 2(M)x4(N), each wave 64x32
//   -> 8 accumulators / 24 WMMAs per 32-K step per wave.
//   Software pipelined: G2R loads for step k+1 issued before compute of k;
//   global_prefetch for step k+2 (one lane per 64B line).
//   mv[i][j] stashed at out[i*T*D + j]  (t=0 slice used as scratch).
// ---------------------------------------------------------------------------
__global__ __launch_bounds__(256)
void gemm_mv_kernel(const float* __restrict__ V,   // (B,D)
                    const float* __restrict__ W,   // (D,D), mv[i][j]=dot(V[i,:],W[j,:])
                    float* __restrict__ out) {
    __shared__ unsigned int sA_hi[128 * 16];
    __shared__ unsigned int sA_lo[128 * 16];
    __shared__ unsigned int sW_hi[128 * 16];
    __shared__ unsigned int sW_lo[128 * 16];

    const int tid    = threadIdx.x;
    const int blockM = blockIdx.y * 128;
    const int blockN = blockIdx.x * 128;

    const int lane = tid & 31;
    const int wave = tid >> 5;
    const int wm   = wave >> 2;  // 0..1 -> M sub-block of 64
    const int wn   = wave & 3;   // 0..3 -> N sub-block of 32
    const int half = lane >> 4;
    const int l15  = lane & 15;

    // per-thread staging coordinates (128x32 tile = 1024 float4, 4/thread)
    int lrow[4], lkq[4];
    size_t aBase[4], bBase[4];
#pragma unroll
    for (int i = 0; i < 4; ++i) {
        int slot = tid + 256 * i;
        lrow[i] = slot >> 3;     // 0..127
        lkq[i]  = slot & 7;      // float4 within 32-wide K row
        aBase[i] = (size_t)(blockM + lrow[i]) * Dn + lkq[i] * 4;
        bBase[i] = (size_t)(blockN + lrow[i]) * Dn + lkq[i] * 4;
    }

    v8f acc[4][2];
    const v8f vzero = {0.f, 0.f, 0.f, 0.f, 0.f, 0.f, 0.f, 0.f};
#pragma unroll
    for (int mi = 0; mi < 4; ++mi)
#pragma unroll
        for (int ni = 0; ni < 2; ++ni) acc[mi][ni] = vzero;

    // prologue: G2R for k-step 0
    float4 pa[4], pb[4];
#pragma unroll
    for (int i = 0; i < 4; ++i) {
        pa[i] = *reinterpret_cast<const float4*>(&V[aBase[i]]);
        pb[i] = *reinterpret_cast<const float4*>(&W[bBase[i]]);
    }

    for (int ks = 0; ks < KSTEPS; ++ks) {
        __syncthreads();  // previous compute finished reading LDS

        // ---- R2S: split f32 -> hi/lo bf16 and stage to LDS ----
#pragma unroll
        for (int i = 0; i < 4; ++i) {
            unsigned int h0, l0, h1, l1;
            const int u = lrow[i] * 16 + lkq[i] * 2;
            split2(pa[i].x, pa[i].y, h0, l0);
            split2(pa[i].z, pa[i].w, h1, l1);
            sA_hi[u] = h0; sA_hi[u + 1] = h1;
            sA_lo[u] = l0; sA_lo[u + 1] = l1;
            split2(pb[i].x, pb[i].y, h0, l0);
            split2(pb[i].z, pb[i].w, h1, l1);
            sW_hi[u] = h0; sW_hi[u + 1] = h1;
            sW_lo[u] = l0; sW_lo[u + 1] = l1;
        }
        __syncthreads();

        // ---- G2R prefetch of k-step ks+1 (waited only at next R2S) ----
        if (ks + 1 < KSTEPS) {
            const int kn = (ks + 1) * 32;
#pragma unroll
            for (int i = 0; i < 4; ++i) {
                pa[i] = *reinterpret_cast<const float4*>(&V[aBase[i] + kn]);
                pb[i] = *reinterpret_cast<const float4*>(&W[bBase[i] + kn]);
            }
        }
        // ---- cache prefetch of k-step ks+2, one lane per 64B line ----
        if ((ks + 2 < KSTEPS) && ((tid & 3) == 0)) {
            const int k2 = (ks + 2) * 32;
#pragma unroll
            for (int i = 0; i < 4; ++i) {
                __builtin_prefetch(&V[aBase[i] + k2], 0, 1);
                __builtin_prefetch(&W[bBase[i] + k2], 0, 1);
            }
        }

        // ---- fragments ----
        v16bf aH[4], aL[4], bH[2], bL[2];
#pragma unroll
        for (int mi = 0; mi < 4; ++mi) {
            const int row = wm * 64 + mi * 16 + l15;
            const int off = row * 16 + half * 4;
            aH[mi] = load_frag(&sA_hi[off]);
            aL[mi] = load_frag(&sA_lo[off]);
        }
#pragma unroll
        for (int ni = 0; ni < 2; ++ni) {
            const int n   = wn * 32 + ni * 16 + l15;
            const int off = n * 16 + half * 4;
            bH[ni] = load_frag(&sW_hi[off]);
            bL[ni] = load_frag(&sW_lo[off]);
        }

        // ---- bf16x3 split product: AB ~= Ahi*Bhi + Ahi*Blo + Alo*Bhi ----
#pragma unroll
        for (int mi = 0; mi < 4; ++mi)
#pragma unroll
            for (int ni = 0; ni < 2; ++ni) {
                acc[mi][ni] = wmma_bf16(aL[mi], bH[ni], acc[mi][ni]);
                acc[mi][ni] = wmma_bf16(aH[mi], bL[ni], acc[mi][ni]);
                acc[mi][ni] = wmma_bf16(aH[mi], bH[ni], acc[mi][ni]);
            }
    }

    // ---- epilogue: D-matrix layout -> out[row*T*D + col] (t=0 scratch) ----
#pragma unroll
    for (int mi = 0; mi < 4; ++mi)
#pragma unroll
        for (int ni = 0; ni < 2; ++ni)
#pragma unroll
            for (int r = 0; r < 8; ++r) {
                const int row = blockM + wm * 64 + mi * 16 + half * 8 + r;
                const int col = blockN + wn * 32 + ni * 16 + l15;
                out[(size_t)row * (Tn * Dn) + col] = acc[mi][ni][r];
            }
}

// ---------------------------------------------------------------------------
// Kernel 2: per-row Poincare dynamics.
//   out[i,t-1,:] = alpha_t*z0 + betam_t*mv + betab_t*b   (all scalars from
//   7 per-row reductions), including both _project() clips.
// ---------------------------------------------------------------------------
__global__ __launch_bounds__(256)
void poincare_rows_kernel(const float* __restrict__ z0,
                          const float* __restrict__ V,
                          const float* __restrict__ bvec,
                          const float* __restrict__ step_sizes,
                          float* __restrict__ out) {
    const int i   = blockIdx.x;
    const int tid = threadIdx.x;
    const size_t rowOut = (size_t)i * Tn * Dn;
    const size_t rowIn  = (size_t)i * Dn;

    float zr[8], mr[8], br[8];
    float s0 = 0.f, s1 = 0.f, s2 = 0.f, s3 = 0.f, s4 = 0.f, s5 = 0.f, s6 = 0.f;

#pragma unroll
    for (int u = 0; u < 8; ++u) {
        int j = tid + 256 * u;
        float zv = z0[rowIn + j];
        float vv = V[rowIn + j];
        float mv = out[rowOut + j];       // mv stashed at t=0 slice
        float bv = bvec[j];
        zr[u] = zv; mr[u] = mv; br[u] = bv;
        s0 += vv * vv;   // ||v||^2
        s1 += mv * mv;   // ||mv||^2
        s2 += mv * bv;   // mv.b
        s3 += zv * zv;   // ||z||^2
        s4 += zv * mv;   // z.mv
        s5 += zv * bv;   // z.b
        s6 += bv * bv;   // ||b||^2
    }

    __shared__ float red[7][256];
    red[0][tid] = s0; red[1][tid] = s1; red[2][tid] = s2; red[3][tid] = s3;
    red[4][tid] = s4; red[5][tid] = s5; red[6][tid] = s6;
    __syncthreads();
    for (int st = 128; st > 0; st >>= 1) {
        if (tid < st) {
#pragma unroll
            for (int q = 0; q < 7; ++q) red[q][tid] += red[q][tid + st];
        }
        __syncthreads();
    }
    const float Svv = red[0][0], Smm = red[1][0], Smb = red[2][0];
    const float Szz = red[3][0], Szm = red[4][0], Szb = red[5][0];
    const float Sbb = red[6][0];

    // ---- scalar chain (identical on every thread; no divergence) ----
    const float vn  = fmaxf(sqrtf(Svv), EPSF);
    const float mvn = fmaxf(sqrtf(Smm), EPSF);
    const float va  = fminf(vn, CLIPAT);
    const float art = 0.5f * logf((1.f + va) / (1.f - va));   // artanh
    const float tr  = tanhf(mvn / vn * art);                  // ||res||
    const float r   = tr / mvn;                               // res = r*mv
    const float bnn = fmaxf(sqrtf(Sbb), EPSF);
    const float sb  = tanhf(bnn) / bnn;                       // b_pt = sb*b

    // mobius_add(res, b_pt) -> v0 = A*mv + C*b, then _project
    const float x2  = tr * tr;
    const float y2  = sb * sb * Sbb;
    const float xy  = r * sb * Smb;
    const float den = fmaxf(1.f + 2.f * xy + x2 * y2, EPSF);
    float A = (1.f + 2.f * xy + y2) * r / den;
    float C = (1.f - x2) * sb / den;
    float n0 = fmaxf(sqrtf(A * A * Smm + 2.f * A * C * Smb + C * C * Sbb), EPSF);
    const float f0 = (n0 > MAXN) ? (MAXN / n0) : 1.f;
    A *= f0; C *= f0;
    const float n0p = n0 * f0;                                // ||v0||

    const float lam  = 2.f / fmaxf(1.f - Szz, EPSF);
    const float step = 1.f / (1.f + expf(-step_sizes[0]));

    for (int t = 1; t <= Tn; ++t) {
        const float c   = step * (float)t;                    // v_t = c*v0
        const float vtn = fmaxf(c * n0p, EPSF);
        const float th  = tanhf(0.5f * lam * vtn);
        const float ss  = th * c / vtn;                       // second = ss*v0
        const float P = A * ss, Q = C * ss;                   // second = P*mv+Q*b
        const float yy  = P * P * Smm + 2.f * P * Q * Smb + Q * Q * Sbb;
        const float xyt = P * Szm + Q * Szb;
        const float dd  = fmaxf(1.f + 2.f * xyt + Szz * yy, EPSF);
        float al = (1.f + 2.f * xyt + yy) / dd;
        float bm = (1.f - Szz) * P / dd;
        float bb = (1.f - Szz) * Q / dd;
        const float nsq = al * al * Szz + bm * bm * Smm + bb * bb * Sbb
                        + 2.f * al * bm * Szm + 2.f * al * bb * Szb
                        + 2.f * bm * bb * Smb;
        const float nn = fmaxf(sqrtf(nsq), EPSF);
        const float ff = (nn > MAXN) ? (MAXN / nn) : 1.f;
        al *= ff; bm *= ff; bb *= ff;

        float* po = out + rowOut + (size_t)(t - 1) * Dn;
#pragma unroll
        for (int u = 0; u < 8; ++u) {
            int j = tid + 256 * u;
            po[j] = al * zr[u] + bm * mr[u] + bb * br[u];
        }
    }
}

// ---------------------------------------------------------------------------
// Launch
// ---------------------------------------------------------------------------
extern "C" void kernel_launch(void* const* d_in, const int* in_sizes, int n_in,
                              void* d_out, int out_size, void* d_ws, size_t ws_size,
                              hipStream_t stream) {
    const float* z0 = (const float*)d_in[0];   // (B,D)
    const float* v  = (const float*)d_in[1];   // (B,D)
    const float* W  = (const float*)d_in[2];   // (D,D)
    const float* b  = (const float*)d_in[3];   // (D,)
    const float* ss = (const float*)d_in[4];   // scalar
    float* out = (float*)d_out;                // (B,T,D)

    dim3 gGemm(Dn / 128, Bn / 128);            // (16, 32)
    gemm_mv_kernel<<<gGemm, dim3(256), 0, stream>>>(v, W, out);

    poincare_rows_kernel<<<dim3(Bn), dim3(256), 0, stream>>>(z0, v, b, ss, out);
}
